// OccupancyPooling_41867341201504
// MI455X (gfx1250) — compile-verified, matrix operand order checked by reading, not run on hardware
//
#include <hip/hip_runtime.h>

// ---------------------------------------------------------------------------
// OccupancyPooling on MI455X (gfx1250, wave32)
//   Stage 1: O(N^2) pairwise binning -> 36-bit occupancy mask per agent,
//            expanded to f16 rows [N, 64] (K padded 36->64 with zeros).
//   Stage 2: occ[ N,64 ] @ Wpad[64,128] + b via v_wmma_f32_16x16x32_f16.
// ---------------------------------------------------------------------------

typedef __attribute__((ext_vector_type(16))) _Float16 v16h;
typedef __attribute__((ext_vector_type(8)))  float    v8f;

#define KPAD 64
#define HID  128
#define KW   36   // real K (N_GRID*N_GRID)

// ---------------------------------------------------------------------------
// Kernel 1: per-agent occupancy bit-mask, written as f16 {0,1} rows.
// ---------------------------------------------------------------------------
__global__ __launch_bounds__(256)
void occ_mask_kernel(const float2* __restrict__ obs,
                     _Float16* __restrict__ occ,   // [n, KPAD] f16
                     int n) {
  __shared__ float2 tile[256];
  const int i = blockIdx.x * 256 + threadIdx.x;
  const float2 oi = obs[i];

  unsigned long long mask = 0ull;
  for (int j0 = 0; j0 < n; j0 += 256) {
    __syncthreads();
    tile[threadIdx.x] = obs[j0 + threadIdx.x];
    __syncthreads();
#pragma unroll 8
    for (int jj = 0; jj < 256; ++jj) {
      // rel = (obs_j - obs_i)/0.5 + 3  == (dj-di)*2 + 3
      const float rx = (tile[jj].x - oi.x) * 2.0f + 3.0f;
      const float ry = (tile[jj].y - oi.y) * 2.0f + 3.0f;
      // NaN in either agent fails every compare -> excluded (matches ref).
      const bool ok = (rx >= 0.0f) & (rx < 6.0f) &
                      (ry >= 0.0f) & (ry < 6.0f) &
                      ((j0 + jj) != i);
      if (ok) {
        const int bin = ((int)rx) * 6 + (int)ry;  // trunc == floor (rx,ry >= 0)
        mask |= 1ull << bin;
      }
    }
  }

  // Expand mask -> 64 f16 values (bits 36..63 never set -> zero padding).
  unsigned int* row = (unsigned int*)(occ + (size_t)i * KPAD);
#pragma unroll
  for (int w = 0; w < 32; ++w) {
    const unsigned int lo = ((mask >> (2 * w))     & 1ull) ? 0x3C00u : 0u; // f16 1.0
    const unsigned int hi = ((mask >> (2 * w + 1)) & 1ull) ? 0x3C00u : 0u;
    row[w] = lo | (hi << 16);
  }
}

// ---------------------------------------------------------------------------
// Kernel 2: out[m, :] = occ[m, :] @ W + b via WMMA f16 -> f32.
//   Block: 256 threads = 8 waves. Block tile: 128 (M) x 128 (N), K = 64.
//   Wave w computes rows [w*16, w*16+16) across all 8 n-tiles.
// ---------------------------------------------------------------------------
__global__ __launch_bounds__(256)
void occ_gemm_kernel(const _Float16* __restrict__ occ,  // [n, KPAD]
                     const float*    __restrict__ W,    // [36, 128] f32
                     const float*    __restrict__ bvec, // [128] f32
                     float*          __restrict__ out)  // [n, 128] f32
{
  __shared__ __align__(16) _Float16 sOcc[128 * KPAD];  // [m][k], 16 KB
  __shared__ __align__(16) _Float16 sWt [HID * KPAD];  // [n][k] transposed, 16 KB

  const int t  = threadIdx.x;
  const int m0 = blockIdx.x * 128;

  // Zero-fill sWt (covers K padding rows 36..63).
#pragma unroll
  for (int r = 0; r < 16; ++r)                  // 8192 halves = 4096 dwords
    ((unsigned int*)sWt)[t + r * 256] = 0u;
  __syncthreads();

  // W[k][n] (f32) -> sWt[n][k] (f16): per-lane fragment reads become
  // contiguous 16B chunks along K.
  for (int e = t; e < KW * HID; e += 256) {
    const int k = e / HID, nn = e % HID;
    sWt[nn * KPAD + k] = (_Float16)W[e];
  }

  // Stage this block's 128x64 occ tile.
  const unsigned int* occU = (const unsigned int*)(occ + (size_t)m0 * KPAD);
#pragma unroll
  for (int r = 0; r < 16; ++r)
    ((unsigned int*)sOcc)[t + r * 256] = occU[t + r * 256];
  __syncthreads();

  const int lane  = t & 31;
  const int wv    = t >> 5;
  const int lhalf = lane >> 4;   // 0: K 0-7 / 16-23, 1: K 8-15 / 24-31
  const int l16   = lane & 15;
  const int kseg  = lhalf * 8;

  union AV { v16h v; float4 f[2]; };

  // A fragments for this wave's 16 rows, two K-steps (K=0..31, K=32..63).
  AV a0, a1;
  const _Float16* arow = &sOcc[(wv * 16 + l16) * KPAD];
  a0.f[0] = *(const float4*)(arow + kseg);
  a0.f[1] = *(const float4*)(arow + kseg + 16);
  a1.f[0] = *(const float4*)(arow + 32 + kseg);
  a1.f[1] = *(const float4*)(arow + 32 + kseg + 16);

#pragma unroll
  for (int ntile = 0; ntile < 8; ++ntile) {
    const int nn = ntile * 16 + l16;
    AV b0, b1;
    const _Float16* brow = &sWt[nn * KPAD];
    b0.f[0] = *(const float4*)(brow + kseg);
    b0.f[1] = *(const float4*)(brow + kseg + 16);
    b1.f[0] = *(const float4*)(brow + 32 + kseg);
    b1.f[1] = *(const float4*)(brow + 32 + kseg + 16);

    v8f c = {};
    c = __builtin_amdgcn_wmma_f32_16x16x32_f16(false, a0.v, false, b0.v,
                                               (short)0, c, false, false);
    c = __builtin_amdgcn_wmma_f32_16x16x32_f16(false, a1.v, false, b1.v,
                                               (short)0, c, false, false);

    const float bias = bvec[nn];
    // C/D layout: VGPR r holds M = r + 8*(lane>=16), N = lane&15.
    float* orow = out + (size_t)(m0 + wv * 16 + lhalf * 8) * HID + nn;
#pragma unroll
    for (int r = 0; r < 8; ++r)
      orow[r * HID] = c[r] + bias;
  }
}

// ---------------------------------------------------------------------------
extern "C" void kernel_launch(void* const* d_in, const int* in_sizes, int n_in,
                              void* d_out, int out_size, void* d_ws, size_t ws_size,
                              hipStream_t stream) {
  const float2* obs  = (const float2*)d_in[0];  // [N, 2] f32
  const float*  W    = (const float*) d_in[1];  // [36, 128] f32
  const float*  bvec = (const float*) d_in[2];  // [128] f32
  float*        out  = (float*)d_out;           // [N, 128] f32
  const int n = in_sizes[0] / 2;                // N_AGENTS (8192)

  _Float16* occ = (_Float16*)d_ws;              // [n, 64] f16 = 1 MB scratch

  occ_mask_kernel<<<n / 256, 256, 0, stream>>>(obs, occ, n);
  occ_gemm_kernel<<<n / 128, 256, 0, stream>>>(occ, W, bvec, out);
}
